// WindowAttention_90288802496825
// MI455X (gfx1250) — matrix operand order repaired and checked by reading, not compiled, not run on hardware
//
#include <hip/hip_runtime.h>
#include <hip/hip_bf16.h>

// ---------------------------------------------------------------------------
// Swin window attention, fused per-window, f16 WMMA (v_wmma_f32_16x16x32_f16)
// + async global->LDS staging of the x tile (ASYNCcnt path)
// ---------------------------------------------------------------------------

constexpr int NH    = 12;     // heads
constexpr int HD    = 32;     // head dim
constexpr int NTK   = 49;     // tokens per window
constexpr int NP    = 64;     // padded tokens (4 x 16)
constexpr int DIMC  = 384;    // channels
constexpr int C3    = 1152;   // 3*DIMC
constexpr int NWIN  = 64;     // windows per image (mask variants)
constexpr int NBLK  = 4096;   // total windows
constexpr float QSCALE = 0.17677669529663687f; // 1/sqrt(32)

typedef _Float16 v16h __attribute__((ext_vector_type(16)));
typedef _Float16 h4v  __attribute__((ext_vector_type(4)));
typedef float    v8f  __attribute__((ext_vector_type(8)));

union AF { v16h h; unsigned u[8]; };

__device__ __forceinline__ v8f wmma_f32_16x16x32(const AF& a, const AF& b, v8f c) {
  return __builtin_amdgcn_wmma_f32_16x16x32_f16(false, a.h, false, b.h,
                                                (short)0, c, false, false);
}

// A-matrix (16x32 f16, ISA 7.12.2): lane m=lane&15, group g=lane>>4,
// vgpr v holds K pair base: v<4 -> 2v+8g ; v>=4 -> 16+2(v-4)+8g
__device__ __forceinline__ int kA(int v, int g) {
  return ((v < 4) ? (2 * v) : (16 + 2 * (v - 4))) + 8 * g;
}
// B-matrix (32x16 f16): lane n=lane&15 is the column; same K order as A:
// vgpr v holds pair K = kA(v, g)
__device__ __forceinline__ int kB(int v, int g) { return kA(v, g); }

// Async 16B/lane copy global -> LDS (GLOBAL_LOAD_ASYNC_TO_LDS_B128, ASYNCcnt)
__device__ __forceinline__ void async_g2l_b128(unsigned lds_off, const void* gaddr) {
  asm volatile("global_load_async_to_lds_b128 %0, %1, off"
               :: "v"(lds_off), "v"((unsigned long long)(size_t)gaddr)
               : "memory");
}
__device__ __forceinline__ void wait_asynccnt0() {
  asm volatile("s_wait_asynccnt 0x0" ::: "memory");
}

// --------------------------- prep kernels ----------------------------------

__global__ void prep_weights_f16(const float* __restrict__ qkv_w,
                                 const float* __restrict__ proj_w,
                                 _Float16* __restrict__ wq,
                                 _Float16* __restrict__ wp) {
  int i = blockIdx.x * blockDim.x + threadIdx.x;
  if (i < C3 * DIMC)   wq[i] = (_Float16)qkv_w[i];
  if (i < DIMC * DIMC) wp[i] = (_Float16)proj_w[i];
}

__global__ void prep_bias(const float* __restrict__ table,
                          const int* __restrict__ relidx,
                          float* __restrict__ biasHNN) {
  int i = blockIdx.x * blockDim.x + threadIdx.x;       // over 12*49*49
  if (i < NH * NTK * NTK) {
    int h  = i / (NTK * NTK);
    int rc = i % (NTK * NTK);
    biasHNN[i] = table[relidx[rc] * NH + h];
  }
}

// --------------------------- main fused kernel ------------------------------

__global__ __launch_bounds__(128)
void swin_win_attn(const float* __restrict__ x,
                   const float* __restrict__ mask,
                   const float* __restrict__ qkv_b,
                   const float* __restrict__ proj_b,
                   const _Float16* __restrict__ wq,     // [1152][384] f16
                   const _Float16* __restrict__ wp,     // [384][384]  f16
                   const float* __restrict__ biasHNN,   // [12][49][49] f32
                   float* __restrict__ out) {
  extern __shared__ __align__(16) char smem[];
  _Float16* qkvS = reinterpret_cast<_Float16*>(smem);                  // 64x1152 f16 (144 KB)
  float*    scS  = reinterpret_cast<float*>(smem + NP * C3 * 2);       // 4 x 64x64 f32 (64 KB)
  _Float16* pxS  = reinterpret_cast<_Float16*>(smem + NP * C3 * 2 + 4 * NP * NP * 4); // x f16 / P f16 (48 KB)
  _Float16* outS = reinterpret_cast<_Float16*>(smem + NP * C3 * 2 + 4 * NP * NP * 4
                                               + NP * DIMC * 2);       // 64x384 f16 (48 KB)
  float* stageF32 = reinterpret_cast<float*>(smem);  // reuse qkv region: 49x384 f32 (75 KB)

  const int tid  = threadIdx.x;
  const int wave = tid >> 5;
  const int lane = tid & 31;
  const int lm   = lane & 15;   // C/D column, A row-in-half, B column
  const int lg   = lane >> 4;   // half-wave group
  const int blk  = blockIdx.x;

  // Warm L2 with the f16 weights (global_prefetch_b8); L2-resident afterwards.
  if (blk == 0) {
    const char* wqb = (const char*)wq;  // 884736 + 294912 bytes, contiguous in ws
    for (int off = tid * 4096; off < (C3 * DIMC + DIMC * DIMC) * 2; off += 128 * 4096)
      __builtin_prefetch(wqb + off, 0, 1);
  }

  // ---- Stage A: async DMA x (49x384 f32) -> LDS staging, then f16 + pad ---
  const float* xw = x + (size_t)blk * NTK * DIMC;
  {
    const unsigned stage_base = (unsigned)(size_t)stageF32;   // LDS byte offset
    for (int i = tid; i < NTK * DIMC / 4; i += 128)           // 4704 x 16B chunks
      async_g2l_b128(stage_base + (unsigned)i * 16u, xw + i * 4);
    wait_asynccnt0();
  }
  __syncthreads();
  for (int i = tid; i < NP * DIMC / 4; i += 128) {
    int r = i / (DIMC / 4);
    int c = (i % (DIMC / 4)) * 4;
    float4 f = make_float4(0.f, 0.f, 0.f, 0.f);
    if (r < NTK) f = *reinterpret_cast<const float4*>(stageF32 + r * DIMC + c);
    h4v hv = {(_Float16)f.x, (_Float16)f.y, (_Float16)f.z, (_Float16)f.w};
    *reinterpret_cast<h4v*>(pxS + r * DIMC + c) = hv;
  }
  __syncthreads();   // all stageF32 reads done before any qkvS write below

  // ---- Stage B: QKV = Xf16 @ Wqkv^T + b  -> qkvS (64x1152 f16) ------------
  {
    const int mt = wave;                       // each wave owns one 16-row block
    for (int nt = 0; nt < C3 / 16; ++nt) {
      const int oc = nt * 16;
      v8f acc = {0.f, 0.f, 0.f, 0.f, 0.f, 0.f, 0.f, 0.f};
      for (int ks = 0; ks < DIMC / 32; ++ks) {
        const int k0 = ks * 32;
        AF a, b;
        #pragma unroll
        for (int v = 0; v < 8; ++v) {
          a.u[v] = *reinterpret_cast<const unsigned*>(pxS + (mt * 16 + lm) * DIMC + k0 + kA(v, lg));
          b.u[v] = *reinterpret_cast<const unsigned*>(wq  + (oc + lm) * DIMC + k0 + kB(v, lg));
        }
        acc = wmma_f32_16x16x32(a, b, acc);
      }
      const float bb = qkv_b[oc + lm];
      #pragma unroll
      for (int r = 0; r < 8; ++r) {
        const int mm = mt * 16 + r + 8 * lg;
        qkvS[mm * C3 + oc + lm] = (_Float16)(acc[r] + bb);
      }
    }
  }
  __syncthreads();

  // ---- Stage C: per-head attention ----------------------------------------
  float*    sc = scS + wave * NP * NP;   // per-wave 64x64 f32 scores
  _Float16* pw = pxS + wave * NP * NP;   // per-wave 64x64 f16 P (reuse x area)
  const float* mrow = mask + (size_t)(blk % NWIN) * NTK * NTK;

  for (int h = wave; h < NH; h += 4) {
    const int qo = h * HD, ko = DIMC + h * HD, vo = 2 * DIMC + h * HD;

    // S = scale * q @ k^T  (K = 32 = hd, one WMMA per 16x16 tile)
    for (int mt = 0; mt < 4; ++mt) {
      for (int nt = 0; nt < 4; ++nt) {
        AF a, b;
        #pragma unroll
        for (int v = 0; v < 8; ++v) {
          a.u[v] = *reinterpret_cast<const unsigned*>(qkvS + (mt * 16 + lm) * C3 + qo + kA(v, lg));
          b.u[v] = *reinterpret_cast<const unsigned*>(qkvS + (nt * 16 + lm) * C3 + ko + kB(v, lg));
        }
        v8f acc = {0.f, 0.f, 0.f, 0.f, 0.f, 0.f, 0.f, 0.f};
        acc = wmma_f32_16x16x32(a, b, acc);
        #pragma unroll
        for (int r = 0; r < 8; ++r)
          sc[(mt * 16 + r + 8 * lg) * NP + nt * 16 + lm] = acc[r] * QSCALE;
      }
    }
    __syncthreads();

    // softmax over real 49 columns, + rel-pos bias + window mask -> P (f16)
    const float* brow = biasHNN + h * NTK * NTK;
    for (int r = lane; r < NP; r += 32) {
      if (r < NTK) {
        float mx = -1e30f;
        for (int j = 0; j < NTK; ++j) {
          float s = sc[r * NP + j] + brow[r * NTK + j] + mrow[r * NTK + j];
          sc[r * NP + j] = s;
          mx = fmaxf(mx, s);
        }
        float sum = 0.f;
        for (int j = 0; j < NTK; ++j) {
          float e = __expf(sc[r * NP + j] - mx);
          sc[r * NP + j] = e;
          sum += e;
        }
        const float inv = 1.f / sum;
        for (int j = 0; j < NTK; ++j) pw[r * NP + j] = (_Float16)(sc[r * NP + j] * inv);
        for (int j = NTK; j < NP; ++j) pw[r * NP + j] = (_Float16)0.f;
      } else {
        for (int j = 0; j < NP; ++j) pw[r * NP + j] = (_Float16)0.f;
      }
    }
    __syncthreads();

    // O = P @ V  (K = 64 -> two WMMA k-steps; 64x32 output per head)
    for (int mt = 0; mt < 4; ++mt) {
      for (int nt = 0; nt < 2; ++nt) {
        v8f acc = {0.f, 0.f, 0.f, 0.f, 0.f, 0.f, 0.f, 0.f};
        for (int ks = 0; ks < 2; ++ks) {
          AF a, b;
          #pragma unroll
          for (int v = 0; v < 8; ++v) {
            a.u[v] = *reinterpret_cast<const unsigned*>(pw + (mt * 16 + lm) * NP + ks * 32 + kA(v, lg));
            #pragma unroll
            for (int j = 0; j < 2; ++j) {
              const int kk = ks * 32 + kB(v, lg) + j;      // V rows stride C3: scalar f16 loads
              b.h[2 * v + j] = qkvS[kk * C3 + vo + nt * 16 + lm];
            }
          }
          acc = wmma_f32_16x16x32(a, b, acc);
        }
        #pragma unroll
        for (int r = 0; r < 8; ++r)
          outS[(mt * 16 + r + 8 * lg) * DIMC + h * HD + nt * 16 + lm] = (_Float16)acc[r];
      }
    }
    __syncthreads();
  }

  // ---- Stage D: out = attnOut @ Wproj^T + b  -> global f32 ----------------
  {
    const int mt = wave;
    float* ow = out + (size_t)blk * NTK * DIMC;
    for (int nt = 0; nt < DIMC / 16; ++nt) {
      const int oc = nt * 16;
      v8f acc = {0.f, 0.f, 0.f, 0.f, 0.f, 0.f, 0.f, 0.f};
      for (int ks = 0; ks < DIMC / 32; ++ks) {
        const int k0 = ks * 32;
        AF a, b;
        #pragma unroll
        for (int v = 0; v < 8; ++v) {
          a.u[v] = *reinterpret_cast<const unsigned*>(outS + (mt * 16 + lm) * DIMC + k0 + kA(v, lg));
          b.u[v] = *reinterpret_cast<const unsigned*>(wp   + (oc + lm) * DIMC + k0 + kB(v, lg));
        }
        acc = wmma_f32_16x16x32(a, b, acc);
      }
      const float bb = proj_b[oc + lm];
      #pragma unroll
      for (int r = 0; r < 8; ++r) {
        const int mm = mt * 16 + r + 8 * lg;
        if (mm < NTK) ow[mm * DIMC + oc + lm] = acc[r] + bb;
      }
    }
  }
}

// --------------------------- launcher --------------------------------------

extern "C" void kernel_launch(void* const* d_in, const int* in_sizes, int n_in,
                              void* d_out, int out_size, void* d_ws, size_t ws_size,
                              hipStream_t stream) {
  const float* x          = (const float*)d_in[0];
  const float* mask       = (const float*)d_in[1];
  const float* qkv_w      = (const float*)d_in[2];
  const float* qkv_b      = (const float*)d_in[3];
  const float* proj_w     = (const float*)d_in[4];
  const float* proj_b     = (const float*)d_in[5];
  const float* bias_table = (const float*)d_in[6];
  const int*   rel_index  = (const int*)d_in[7];
  float* out = (float*)d_out;

  char* ws = (char*)d_ws;
  _Float16* wq      = (_Float16*)ws;                            // 1152*384 f16
  _Float16* wp      = (_Float16*)(ws + C3 * DIMC * 2);          // 384*384  f16
  float*    biasHNN = (float*)(ws + C3 * DIMC * 2 + DIMC * DIMC * 2); // 12*49*49 f32

  {
    int n = C3 * DIMC;
    prep_weights_f16<<<(n + 255) / 256, 256, 0, stream>>>(qkv_w, proj_w, wq, wp);
  }
  {
    int n = NH * NTK * NTK;
    prep_bias<<<(n + 255) / 256, 256, 0, stream>>>(bias_table, rel_index, biasHNN);
  }

  const size_t shmem = (size_t)NP * C3 * 2      // qkv f16 (aliased by f32 x staging)
                     + (size_t)4 * NP * NP * 4  // per-wave scores f32
                     + (size_t)NP * DIMC * 2    // x / P f16
                     + (size_t)NP * DIMC * 2;   // attn out f16  => 311296 B
  swin_win_attn<<<NBLK, 128, shmem, stream>>>(x, mask, qkv_b, proj_b,
                                              wq, wp, biasHNN, out);
}
